// MsgModelDiff_56916906607069
// MI455X (gfx1250) — matrix operand order repaired and checked by reading, not compiled, not run on hardware
//
#include <hip/hip_runtime.h>
#include <hip/hip_bf16.h>

// ---------------------------------------------------------------------------
// GNN message-passing on MI455X (gfx1250, wave32).
// Edge MLP (the ~125 GFLOP hot path) runs on v_wmma_f32_16x16x32_f16:
//   A[16x64] x Wm1[64x224] -> relu -> x Wm2[224x32] -> *(xi-xj) -> x W2[32x32]
// f16 inputs / f32 accumulation; weights fragment-swizzled in LDS so B-frag
// loads are contiguous b128 pairs; A fragments register-blocked across
// n-tiles. segment_sum via global f32 atomics.
// ---------------------------------------------------------------------------

typedef _Float16 h16;
typedef __attribute__((ext_vector_type(16))) _Float16 v16h;
typedef __attribute__((ext_vector_type(8)))  _Float16 v8h;
typedef __attribute__((ext_vector_type(8)))  float    v8f;

#define MSG   200
#define MSGP  224            // padded msg dim: 14 n-tiles of 16, 7 k-blocks of 32
#define NT1   (MSGP / 16)    // 14
#define K2B   (MSGP / 32)    // 7
#define CP    32             // padded c_in / c_out
#define WPB   4              // waves per block
#define TPB   (WPB * 32)

// -------------------- LDS sizing (host + device share this) ----------------
__host__ __device__ constexpr int edge_k1b(int cin) { return (2 * cin + 2 + 31) / 32; }
__host__ __device__ constexpr size_t edge_smem_bytes(int cin) {
    const int K1B = edge_k1b(cin);
    const int K1P = 32 * K1B;
    size_t weights_h = (size_t)K1B * NT1 * 512 + (size_t)K2B * 2 * 512 + 2 * 512; // halves
    size_t bias_b    = 4 * (MSGP + CP + CP);
    size_t wave_b    = (size_t)2 * (16 * K1P) + 2 * (16 * MSGP) + 4 * (16 * CP) + 4 * 32;
    return weights_h * 2 + bias_b + (size_t)WPB * wave_b;
}

// ------------- cooperative store of W[Kreal][Nreal] (f32, row-major) -------
// into fragment-swizzled padded f16: block(kb,nt) holds 512 halves laid out
// [lane][j] so a B fragment is 32 contiguous bytes per lane.
__device__ inline void stage_b(const float* __restrict__ W, h16* __restrict__ dst,
                               int Kreal, int Nreal, int KBLK, int NTN,
                               int tid, int nthreads) {
    int total = KBLK * NTN * 512;
    for (int idx = tid; idx < total; idx += nthreads) {
        int j    = idx & 15;
        int lane = (idx >> 4) & 31;
        int blk  = idx >> 9;
        int kb   = blk / NTN, nt = blk % NTN;
        int k = kb * 32 + (j >> 3) * 16 + (lane >> 4) * 8 + (j & 7);   // B frag: k = kbase + (j>>3)*16 + hi*8 + (j&7)
        int n = nt * 16 + (lane & 15);            //         col = nb + (lane&15)
        float v = (k < Kreal && n < Nreal) ? W[k * Nreal + n] : 0.0f;
        dst[idx] = (h16)v;
    }
}

// A fragment from a row-major f16 LDS tile (ld halves per row, 16B aligned).
// ISA 16-bit A layout: lane m=lane&15, hi=lane>>4;
//   element j -> K = (j>>3)*16 + hi*8 + (j&7)   (relative to kbase)
__device__ inline v16h load_a(const h16* __restrict__ tile, int ld, int kbase,
                              int m, int hi) {
    const h16* p = tile + m * ld + kbase + hi * 8;
    v8h lo = *(const v8h*)(p);
    v8h hh = *(const v8h*)(p + 16);
    v16h a;
#pragma unroll
    for (int j = 0; j < 8; ++j) { a[j] = lo[j]; a[j + 8] = hh[j]; }
    return a;
}

// ------------------------------- edge kernel -------------------------------
template <int CIN, int COUT>
__global__ __launch_bounds__(TPB)
void edge_kernel(const float* __restrict__ x,
                 const int* __restrict__ srcI, const int* __restrict__ dstI,
                 const float* __restrict__ ew,
                 const float* __restrict__ Wm1, const float* __restrict__ bm1,
                 const float* __restrict__ Wm2, const float* __restrict__ bm2,
                 const float* __restrict__ W2,  const float* __restrict__ b2,
                 float* __restrict__ agg, int E) {
    constexpr int KREAL = 2 * CIN + 2;
    constexpr int K1B = (KREAL + 31) / 32;
    constexpr int K1P = 32 * K1B;
    constexpr int NT3 = (COUT > 16) ? 2 : 1;
    constexpr int AW  = 16 * K1P;    // halves
    constexpr int HWH = 16 * MSGP;   // halves
    constexpr int DWF = 16 * CP;     // floats
    constexpr int WAVE_BYTES = AW * 2 + HWH * 2 + DWF * 4 + 32 * 4;

    extern __shared__ char smem_raw[];
    h16*   swzWm1 = (h16*)smem_raw;                      // K1B*NT1*512 halves
    h16*   swzWm2 = swzWm1 + K1B * NT1 * 512;            // K2B*2*512
    h16*   swzW2  = swzWm2 + K2B * 2 * 512;              // 2*512
    float* bm1s   = (float*)(swzW2 + 2 * 512);           // MSGP
    float* bm2s   = bm1s + MSGP;                         // CP
    float* b2s    = bm2s + CP;                           // CP
    char*  waveB0 = (char*)(b2s + CP);

    const int tid  = threadIdx.x;
    const int wave = tid >> 5;
    const int lane = tid & 31;

    char* wb = waveB0 + (size_t)wave * WAVE_BYTES;
    h16*   aTile = (h16*)wb;
    h16*   hTile = aTile + AW;
    float* dTile = (float*)(hTile + HWH);
    int*   idxT  = (int*)(dTile + DWF);                  // [0..15]=dst, [16..31]=src
    h16*   gTile = aTile;                                // reuse (aTile dead post-GEMM1)

    // ---- stage weights/biases into LDS once per block ----
    stage_b(Wm1, swzWm1, KREAL, MSG,  K1B, NT1, tid, blockDim.x);
    stage_b(Wm2, swzWm2, MSG,   CIN,  K2B, 2,   tid, blockDim.x);
    stage_b(W2,  swzW2,  CIN,   COUT, 1,   2,   tid, blockDim.x);
    for (int i = tid; i < MSGP; i += blockDim.x) bm1s[i] = (i < MSG) ? bm1[i] : 0.f;
    for (int i = tid; i < CP; i += blockDim.x) {
        bm2s[i] = (i < CIN)  ? bm2[i] : 0.f;
        b2s[i]  = (i < COUT) ? b2[i]  : 0.f;
    }
    __syncthreads();

    const int m  = lane & 15;
    const int hi = lane >> 4;
    const int numTiles = (E + 15) >> 4;
    const int gw = blockIdx.x * WPB + wave;
    const int nw = gridDim.x * WPB;

    for (int t = gw; t < numTiles; t += nw) {
        const int e0 = t << 4;

        // stage edge indices for this 16-edge tile
        {
            int e = e0 + m;
            int idx = (e < E) ? ((hi == 0) ? dstI[e] : srcI[e]) : 0;
            idxT[lane] = idx;
        }
        // build A = [xi | xj | ea | 0] (f16) and D = xi - xj (f32)
        for (int idx = lane; idx < 16 * K1P; idx += 32) {
            int er = idx / K1P, k = idx % K1P;
            int e = e0 + er;
            float v = 0.f;
            if (e < E) {
                if (k < CIN)            v = x[idxT[er] * CIN + k];
                else if (k < 2 * CIN)   v = x[idxT[16 + er] * CIN + (k - CIN)];
                else if (k < KREAL)     v = ew[e * 2 + (k - 2 * CIN)];
            }
            aTile[idx] = (h16)v;
        }
        for (int idx = lane; idx < 16 * CP; idx += 32) {
            int er = idx >> 5, c = idx & (CP - 1);
            int e = e0 + er;
            float v = 0.f;
            if (e < E && c < CIN)
                v = x[idxT[er] * CIN + c] - x[idxT[16 + er] * CIN + c];
            dTile[idx] = v;
        }
        // per-wave LDS producer/consumer: DS ops are in-order within a wave.

        // ---- GEMM1: h = relu(A @ Wm1 + bm1) -> hTile (f16) ----
        // A fragments register-blocked: loaded once, reused by all 14 n-tiles.
        v16h aF[K1B];
#pragma unroll
        for (int kb = 0; kb < K1B; ++kb) aF[kb] = load_a(aTile, K1P, kb * 32, m, hi);

        for (int nt = 0; nt < NT1; ++nt) {
            v8f acc = {};
#pragma unroll
            for (int kb = 0; kb < K1B; ++kb) {
                v16h b = *(const v16h*)(swzWm1 + (((kb * NT1 + nt) << 9) + lane * 16));
                acc = __builtin_amdgcn_wmma_f32_16x16x32_f16(
                          false, aF[kb], false, b, (short)0, acc, false, false);
            }
            int col = nt * 16 + m;
            float bias = bm1s[col];
#pragma unroll
            for (int r = 0; r < 8; ++r) {
                float v = acc[r] + bias;          // C/D: row = r + hi*8, col = lane&15
                v = v > 0.f ? v : 0.f;
                hTile[(r + hi * 8) * MSGP + col] = (h16)v;
            }
        }

        // ---- GEMM2: g = (h @ Wm2 + bm2) * (xi - xj) -> gTile (f16) ----
        // Both n-tile accumulators live across K so each A frag loads once.
        {
            v8f acc0 = {}, acc1 = {};
#pragma unroll
            for (int kb = 0; kb < K2B; ++kb) {
                v16h a  = load_a(hTile, MSGP, kb * 32, m, hi);
                v16h b0 = *(const v16h*)(swzWm2 + (((kb * 2 + 0) << 9) + lane * 16));
                v16h b1 = *(const v16h*)(swzWm2 + (((kb * 2 + 1) << 9) + lane * 16));
                acc0 = __builtin_amdgcn_wmma_f32_16x16x32_f16(
                           false, a, false, b0, (short)0, acc0, false, false);
                acc1 = __builtin_amdgcn_wmma_f32_16x16x32_f16(
                           false, a, false, b1, (short)0, acc1, false, false);
            }
#pragma unroll
            for (int nt = 0; nt < 2; ++nt) {
                const v8f& acc = nt ? acc1 : acc0;
                int col = nt * 16 + m;
                float bias = bm2s[col];
#pragma unroll
                for (int r = 0; r < 8; ++r) {
                    int row = r + hi * 8;
                    float g = (acc[r] + bias) * dTile[row * CP + col]; // d==0 in pad
                    gTile[row * CP + col] = (h16)g;
                }
            }
        }

        // ---- GEMM3: msg = g @ W2 + b2; atomic segment-sum into agg[dst] ----
        {
            v16h a = load_a(gTile, CP, 0, m, hi);
#pragma unroll
            for (int nt = 0; nt < NT3; ++nt) {
                v8f acc = {};
                v16h b = *(const v16h*)(swzW2 + ((nt << 9) + lane * 16));
                acc = __builtin_amdgcn_wmma_f32_16x16x32_f16(
                          false, a, false, b, (short)0, acc, false, false);
                int col = nt * 16 + m;
                if (col < COUT) {
                    float bias = b2s[col];
#pragma unroll
                    for (int r = 0; r < 8; ++r) {
                        int er = r + hi * 8;
                        int e = e0 + er;
                        if (e < E)
                            atomicAdd(&agg[idxT[er] * COUT + col], acc[r] + bias);
                    }
                }
            }
        }
    }
}

// ------------------------------- node kernel -------------------------------
__global__ void node_kernel(const float* __restrict__ x, const float* __restrict__ agg,
                            const float* __restrict__ W1, const float* __restrict__ b1,
                            float* __restrict__ out, int N, int CIN, int COUT,
                            int doRelu) {
    int i = blockIdx.x * blockDim.x + threadIdx.x;
    if (i >= N * COUT) return;
    int n = i / COUT, c = i % COUT;
    float s = agg[i] + b1[c];
    for (int k = 0; k < CIN; ++k) s += x[n * CIN + k] * W1[k * COUT + c];
    if (doRelu) s = s > 0.f ? s : 0.f;
    out[i] = s;
}

__global__ void zero_kernel(float* __restrict__ p, int n) {
    int i = blockIdx.x * blockDim.x + threadIdx.x;
    if (i < n) p[i] = 0.f;
}

// ------------------------------- launcher ----------------------------------
extern "C" void kernel_launch(void* const* d_in, const int* in_sizes, int n_in,
                              void* d_out, int out_size, void* d_ws, size_t ws_size,
                              hipStream_t stream) {
    const float* features = (const float*)d_in[0];
    const int*   edges    = (const int*)d_in[1];
    const float* ew       = (const float*)d_in[2];
    const int E = in_sizes[1] / 2;
    const int N = in_sizes[0];          // NUM_IN == 1
    const int* srcI = edges;
    const int* dstI = edges + E;

    // params: [3..10]=d, [11..18]=h, [19..26]=o; order W1,b1,Wm1,bm1,Wm2,bm2,W2,b2
    auto P = [&](int base, int off) { return (const float*)d_in[base + off]; };

    float* xA  = (float*)d_ws;                 // N*30
    float* xB  = xA + (size_t)N * 30;          // N*30
    float* agg = xB + (size_t)N * 30;          // N*30

    const dim3 eBlk(TPB), eGrd(960);
    const size_t smem1  = edge_smem_bytes(1);
    const size_t smem30 = edge_smem_bytes(30);
    auto cdiv = [](int a, int b) { return (a + b - 1) / b; };

    // ---- layer d: c_in=1 -> c_out=30, relu ----
    zero_kernel<<<cdiv(N * 30, 256), 256, 0, stream>>>(agg, N * 30);
    edge_kernel<1, 30><<<eGrd, eBlk, smem1, stream>>>(
        features, srcI, dstI, ew, P(3, 2), P(3, 3), P(3, 4), P(3, 5), P(3, 6), P(3, 7),
        agg, E);
    node_kernel<<<cdiv(N * 30, 256), 256, 0, stream>>>(
        features, agg, P(3, 0), P(3, 1), xA, N, 1, 30, 1);

    // ---- layer h (shared weights) x3: 30 -> 30, relu ----
    float* cur = xA;
    float* nxt = xB;
    for (int it = 0; it < 3; ++it) {
        zero_kernel<<<cdiv(N * 30, 256), 256, 0, stream>>>(agg, N * 30);
        edge_kernel<30, 30><<<eGrd, eBlk, smem30, stream>>>(
            cur, srcI, dstI, ew, P(11, 2), P(11, 3), P(11, 4), P(11, 5), P(11, 6),
            P(11, 7), agg, E);
        node_kernel<<<cdiv(N * 30, 256), 256, 0, stream>>>(
            cur, agg, P(11, 0), P(11, 1), nxt, N, 30, 30, 1);
        float* t = cur; cur = nxt; nxt = t;
    }

    // ---- layer o: 30 -> 1, no relu ----
    zero_kernel<<<cdiv(N, 256), 256, 0, stream>>>(agg, N);
    edge_kernel<30, 1><<<eGrd, eBlk, smem30, stream>>>(
        cur, srcI, dstI, ew, P(19, 2), P(19, 3), P(19, 4), P(19, 5), P(19, 6), P(19, 7),
        agg, E);
    node_kernel<<<cdiv(N, 256), 256, 0, stream>>>(
        cur, agg, P(19, 0), P(19, 1), (float*)d_out, N, 30, 1, 0);
}
